// RNNEncModel_39419209843221
// MI455X (gfx1250) — matrix operand order repaired and checked by reading, not compiled, not run on hardware
//
#include <hip/hip_runtime.h>
#include <hip/hip_bf16.h>
#include <math.h>

// Problem constants (from reference)
#define BB 64
#define TT 512
#define EE 512
#define HH 1024
#define LL 2
#define BT (BB * TT)

typedef __attribute__((ext_vector_type(16))) _Float16 v16h;
typedef __attribute__((ext_vector_type(8)))  _Float16 v8h;
typedef __attribute__((ext_vector_type(8)))  float    v8f;

// ---------------------------------------------------------------------------
// A-fragment (16x32 f16, row-major, stride ld), per CDNA5 ISA 7.12.2:
//   lane L (<16):  row = L,    K = {0..7} in v[0..7],  {16..23} in v[8..15]
//   lane L (>=16): row = L-16, K = {8..15},             {24..31}
// ---------------------------------------------------------------------------
__device__ __forceinline__ v16h load_fragA(const _Float16* __restrict__ base,
                                           int ld, int row0, int k0) {
    int lane = threadIdx.x & 31;
    int r  = row0 + (lane & 15);
    int kb = k0 + ((lane >> 4) << 3);
    const _Float16* p = base + (size_t)r * ld + kb;
    v8h lo = *(const v8h*)(p);        // K = kb .. kb+7
    v8h hi = *(const v8h*)(p + 16);   // K = kb+16 .. kb+23
    v16h out;
#pragma unroll
    for (int i = 0; i < 8; ++i) { out[i] = lo[i]; out[8 + i] = hi[i]; }
    return out;
}

// ---------------------------------------------------------------------------
// B-fragment (32x16 f16 = K x N). W is stored [N,K] row-major, so column n of
// B is row n of W. Same K order as A-fragment: lane L holds
//   n = L%16, K = (L>=16 ? 8 : 0) + {0..7} in v[0..7], +16 in v[8..15].
// ---------------------------------------------------------------------------
__device__ __forceinline__ v16h load_fragB(const _Float16* __restrict__ base,
                                           int ld, int col0, int k0) {
    int lane = threadIdx.x & 31;
    int n  = col0 + (lane & 15);
    int kb = k0 + ((lane >> 4) << 3);
    const _Float16* p = base + (size_t)n * ld + kb;
    v8h lo = *(const v8h*)(p);
    v8h hi = *(const v8h*)(p + 16);
    v16h out;
#pragma unroll
    for (int i = 0; i < 8; ++i) { out[i] = lo[i]; out[8 + i] = hi[i]; }
    return out;
}

// ---------------------------------------------------------------------------
// f32 -> f16 cast (grid-stride)
// ---------------------------------------------------------------------------
__global__ void cast_f16_kernel(const float* __restrict__ in,
                                _Float16* __restrict__ out, size_t n) {
    size_t i = (size_t)blockIdx.x * blockDim.x + threadIdx.x;
    size_t stride = (size_t)gridDim.x * blockDim.x;
    for (; i < n; i += stride) out[i] = (_Float16)in[i];
}

// ---------------------------------------------------------------------------
// Embedding gather with padding_idx = 0, f16 output [BT, E]
// ---------------------------------------------------------------------------
__global__ void embed_kernel(const int* __restrict__ src,
                             const float* __restrict__ table,
                             _Float16* __restrict__ xe) {
    int row = blockIdx.x;                  // 0 .. BT-1  (= b*T + t)
    int idx = src[row];
    const float* trow = table + (size_t)idx * EE;
    for (int e = threadIdx.x; e < EE; e += blockDim.x)
        xe[(size_t)row * EE + e] = (idx == 0) ? (_Float16)0.0f : (_Float16)trow[e];
}

__global__ void zero_h_kernel(_Float16* __restrict__ h, int n) {
    int i = blockIdx.x * blockDim.x + threadIdx.x;
    if (i < n) h[i] = (_Float16)0.0f;
}

// ---------------------------------------------------------------------------
// WMMA GEMM: out[M,N] = act(A[M,K] * W[N,K]^T + bias)
// OUT_MODE 0: f32 out, no activation (pre-activations for the scan)
// OUT_MODE 1: f16 out, ReLU (MLP stages, feeds the next GEMM)
// Block = 128 threads = 4 waves as 2(M) x 2(N); wave tile = 32(M) x 64(N).
// Block tile = 64(M) x 128(N). 8 WMMAs per wave per K-step on 12 b128 loads.
// ---------------------------------------------------------------------------
template <int OUT_MODE>
__global__ void gemm_wmma_kernel(const _Float16* __restrict__ A,
                                 const _Float16* __restrict__ W,
                                 const float* __restrict__ bias,
                                 void* __restrict__ out,
                                 int M, int N, int K) {
    int wave = threadIdx.x >> 5;
    int lane = threadIdx.x & 31;
    int mw = wave & 1;          // wave position in block: 2 along M
    int nw = wave >> 1;         //                          2 along N
    int row0 = blockIdx.y * 64 + mw * 32;
    int col0 = blockIdx.x * 128 + nw * 64;
    if (row0 >= M) return;   // wave-uniform: surviving waves keep EXEC all-1s

    v8f acc[2][4] = {};
    for (int kk = 0; kk < K; kk += 32) {
        v16h a0 = load_fragA(A, K, row0, kk);
        v16h a1 = load_fragA(A, K, row0 + 16, kk);
#pragma unroll
        for (int j = 0; j < 4; ++j) {
            v16h b = load_fragB(W, K, col0 + j * 16, kk);
            acc[0][j] = __builtin_amdgcn_wmma_f32_16x16x32_f16(
                false, a0, false, b, (short)0, acc[0][j], false, false);
            acc[1][j] = __builtin_amdgcn_wmma_f32_16x16x32_f16(
                false, a1, false, b, (short)0, acc[1][j], false, false);
        }
    }
    // Epilogue. C layout: vgpr r -> m = row0+16i + r + 8*(lane>=16),
    //                              n = col0 + j*16 + (lane&15)
    int moff = (lane >> 4) << 3;
    int nc = lane & 15;
#pragma unroll
    for (int i = 0; i < 2; ++i) {
        int mbase = row0 + i * 16 + moff;
#pragma unroll
        for (int j = 0; j < 4; ++j) {
            int n = col0 + j * 16 + nc;
            float bv = bias[n];
#pragma unroll
            for (int r = 0; r < 8; ++r) {
                float v = acc[i][j][r] + bv;
                size_t idx = (size_t)(mbase + r) * N + n;
                if (OUT_MODE == 1) {
                    v = fmaxf(v, 0.0f);
                    ((_Float16*)out)[idx] = (_Float16)v;
                } else {
                    ((float*)out)[idx] = v;
                }
            }
        }
    }
}

// ---------------------------------------------------------------------------
// One RNN timestep, both directions (blockIdx.z = dir):
//   h_new = tanh(pre[:, t_eff, :] + h_old @ Whh^T + b_hh)
// pre laid out [dir][B*T, H] with row = b*T + t. Writes f16 h state, optional
// concat output cat[b*T+t_eff, dir*H + n], and f32 finals on the last step.
// Narrow 16x64 wave tile to maximize wave count (latency-bound step GEMM).
// Grid: (H/64, 1, 2), block 128 (4 waves covering B=64 rows).
// ---------------------------------------------------------------------------
__global__ void rnn_step_kernel(const _Float16* __restrict__ hin,   // [2][B*H]
                                _Float16* __restrict__ hout,        // [2][B*H]
                                const _Float16* __restrict__ Whh,   // [2][H*H]
                                const float* __restrict__ pre,      // [2][BT*H]
                                const float* __restrict__ bhh,      // [2][H]
                                _Float16* __restrict__ cat,         // [BT, 2H] or null
                                float* __restrict__ finals,         // [2][B*H]
                                int t, int is_last) {
    int dir  = blockIdx.z;
    int wave = threadIdx.x >> 5;
    int lane = threadIdx.x & 31;
    int row0 = wave * 16;           // batch tile
    int col0 = blockIdx.x * 64;     // hidden tile

    const _Float16* hA = hin + (size_t)dir * (BB * HH);
    const _Float16* Wd = Whh + (size_t)dir * HH * HH;

    v8f acc[4] = {};
    for (int kk = 0; kk < HH; kk += 32) {
        v16h a = load_fragA(hA, HH, row0, kk);
#pragma unroll
        for (int j = 0; j < 4; ++j) {
            v16h b = load_fragB(Wd, HH, col0 + j * 16, kk);
            acc[j] = __builtin_amdgcn_wmma_f32_16x16x32_f16(
                false, a, false, b, (short)0, acc[j], false, false);
        }
    }

    int t_eff = (dir == 0) ? t : (TT - 1 - t);
    int mbase = row0 + ((lane >> 4) << 3);
    int nc = lane & 15;
    const float* pd = pre + (size_t)dir * BT * HH;
    _Float16* hod = hout + (size_t)dir * (BB * HH);

#pragma unroll
    for (int j = 0; j < 4; ++j) {
        int n = col0 + j * 16 + nc;
        float bb = bhh[dir * HH + n];
#pragma unroll
        for (int r = 0; r < 8; ++r) {
            int b = mbase + r;
            float v = tanhf(acc[j][r] + bb + pd[((size_t)b * TT + t_eff) * HH + n]);
            hod[(size_t)b * HH + n] = (_Float16)v;
            if (cat)
                cat[((size_t)b * TT + t_eff) * (2 * HH) + dir * HH + n] = (_Float16)v;
            if (is_last)
                finals[(size_t)dir * BB * HH + (size_t)b * HH + n] = v;
        }
    }
}

// ---------------------------------------------------------------------------
// Host launcher
// ---------------------------------------------------------------------------
extern "C" void kernel_launch(void* const* d_in, const int* in_sizes, int n_in,
                              void* d_out, int out_size, void* d_ws, size_t ws_size,
                              hipStream_t stream) {
    const int*   src   = (const int*)  d_in[0];
    const float* emb   = (const float*)d_in[1];
    const float* W1    = (const float*)d_in[2];
    const float* b1    = (const float*)d_in[3];
    const float* W2    = (const float*)d_in[4];
    const float* b2    = (const float*)d_in[5];
    const float* Wih0  = (const float*)d_in[6];
    const float* WihL  = (const float*)d_in[7];
    const float* Whh   = (const float*)d_in[8];
    const float* bih   = (const float*)d_in[9];
    const float* bhh   = (const float*)d_in[10];
    float* out = (float*)d_out;
    char*  ws  = (char*)d_ws;

    // workspace carve-out (256B aligned)
    size_t off = 0;
    auto carve = [&](size_t bytes) -> char* {
        off = (off + 255) & ~(size_t)255;
        char* p = ws + off;
        off += bytes;
        return p;
    };
    _Float16* wW1   = (_Float16*)carve((size_t)HH * EE * 2);          //  1 MB
    _Float16* wW2   = (_Float16*)carve((size_t)HH * HH * 2);          //  2 MB
    _Float16* wWih0 = (_Float16*)carve((size_t)2 * HH * HH * 2);      //  4 MB
    _Float16* wWihL = (_Float16*)carve((size_t)2 * HH * 2 * HH * 2);  //  8 MB
    _Float16* wWhh  = (_Float16*)carve((size_t)4 * HH * HH * 2);      //  8 MB
    _Float16* xe    = (_Float16*)carve((size_t)BT * EE * 2);          // 32 MB
    _Float16* h1    = (_Float16*)carve((size_t)BT * HH * 2);          // 64 MB
    _Float16* h2    = (_Float16*)carve((size_t)BT * HH * 2);          // 64 MB
    _Float16* cat   = (_Float16*)carve((size_t)BT * 2 * HH * 2);      // 128 MB
    float*    pre   = (float*)   carve((size_t)2 * BT * HH * 4);      // 256 MB
    _Float16* hbuf  = (_Float16*)carve((size_t)2 * 2 * BB * HH * 2);  // 0.5 MB

    // 1) weight casts to f16
    cast_f16_kernel<<<1024, 256, 0, stream>>>(W1,   wW1,   (size_t)HH * EE);
    cast_f16_kernel<<<1024, 256, 0, stream>>>(W2,   wW2,   (size_t)HH * HH);
    cast_f16_kernel<<<1024, 256, 0, stream>>>(Wih0, wWih0, (size_t)2 * HH * HH);
    cast_f16_kernel<<<1024, 256, 0, stream>>>(WihL, wWihL, (size_t)2 * HH * 2 * HH);
    cast_f16_kernel<<<1024, 256, 0, stream>>>(Whh,  wWhh,  (size_t)4 * HH * HH);

    // 2) embedding gather (padding_idx = 0)
    embed_kernel<<<BT, 128, 0, stream>>>(src, emb, xe);

    // 3) MLP: h1 = relu(x W1^T + b1); h2 = relu(h1 W2^T + b2)
    gemm_wmma_kernel<1><<<dim3(HH / 128, BT / 64), 128, 0, stream>>>(
        xe, wW1, b1, h1, BT, HH, EE);
    gemm_wmma_kernel<1><<<dim3(HH / 128, BT / 64), 128, 0, stream>>>(
        h1, wW2, b2, h2, BT, HH, HH);

    // 4) RNN layers
    for (int l = 0; l < LL; ++l) {
        const _Float16* Ain = (l == 0) ? h2 : cat;
        int Kin             = (l == 0) ? HH : 2 * HH;
        const _Float16* Wl  = (l == 0) ? wWih0 : wWihL;

        // input projections for both directions: pre[d] = Ain @ Wih[d]^T + bih[l,d]
        for (int d = 0; d < 2; ++d) {
            gemm_wmma_kernel<0><<<dim3(HH / 128, BT / 64), 128, 0, stream>>>(
                Ain, Wl + (size_t)d * HH * Kin, bih + (size_t)(l * 2 + d) * HH,
                pre + (size_t)d * BT * HH, BT, HH, Kin);
        }

        // zero initial hidden state (parity-0 buffer, both dirs)
        zero_h_kernel<<<(2 * BB * HH + 255) / 256, 256, 0, stream>>>(hbuf, 2 * BB * HH);

        _Float16* cat_w = (l < LL - 1) ? cat : nullptr;   // last layer: outputs unused
        float* finals = out + (size_t)l * 2 * BB * HH;

        for (int t = 0; t < TT; ++t) {
            _Float16* hin  = hbuf + (size_t)(t & 1) * 2 * BB * HH;
            _Float16* hout = hbuf + (size_t)((t + 1) & 1) * 2 * BB * HH;
            rnn_step_kernel<<<dim3(HH / 64, 1, 2), 128, 0, stream>>>(
                hin, hout, wWhh + (size_t)l * 2 * HH * HH, pre,
                bhh + (size_t)l * 2 * HH, cat_w, finals, t, (t == TT - 1) ? 1 : 0);
        }
    }
}
